// Matching_fb_76527727280173
// MI455X (gfx1250) — compile-verified, matrix operand order checked by reading, not run on hardware
//
#include <hip/hip_runtime.h>
#include <hip/hip_bf16.h>
#include <math.h>

typedef __attribute__((ext_vector_type(16))) _Float16 v16h;
typedef __attribute__((ext_vector_type(8)))  _Float16 v8h;
typedef __attribute__((ext_vector_type(8)))  float    v8f;

#define PLANES 512
#define HEADS 8
#define DHEAD 64
#define BB 4
#define NN 1024
#define ROWS (BB * NN)          // 4096
#define QKVW (3 * PLANES)       // 1536

// Build a WMMA 16-bit A/B fragment from a K-contiguous LDS row.
// Lane layout (ISA 7.12.2): lanes 0-15 hold K base..base+7 and base+16..base+23
// with base = 8*(lane>=16). Two 16-byte runs -> two ds_load_b128.
__device__ __forceinline__ v16h frag_ld(const _Float16* rowp, int base) {
    v8h lo = *(const v8h*)(rowp + base);
    v8h hi = *(const v8h*)(rowp + base + 16);
    v16h r;
    #pragma unroll
    for (int e = 0; e < 8; ++e) { r[e] = lo[e]; r[e + 8] = hi[e]; }
    return r;
}

// ---------------------------------------------------------------------------
// casts
// ---------------------------------------------------------------------------
__global__ void cast_f16_kernel(const float* __restrict__ in,
                                _Float16* __restrict__ out, int n) {
    int i = blockIdx.x * blockDim.x + threadIdx.x;
    if (i < n) out[i] = (_Float16)in[i];
}

// in: [K][N] f32 -> out: [N][K] f16 (weights, one-time)
__global__ void cast_transpose_kernel(const float* __restrict__ in,
                                      _Float16* __restrict__ out, int K, int N) {
    int idx = blockIdx.x * blockDim.x + threadIdx.x;
    if (idx >= K * N) return;
    int k = idx / N, n = idx - k * N;
    out[(size_t)n * K + k] = (_Float16)in[idx];
}

// ---------------------------------------------------------------------------
// WMMA GEMM, transposed-B convention: C[M,N] = scale*(A[M,K] @ Bt[N,K]^T)
// Block 64(M) x 128(N); 8 waves as 2x4; wave = 32x32 = 2x2 16x16 frags.
// qkvmode: de-interleave heads; Q/K -> Ch [r][sec*512+h*64+d], V -> Vh [b][h][d][n]
// ---------------------------------------------------------------------------
__global__ __launch_bounds__(256) void gemm_f16_kernel(
    const _Float16* __restrict__ A, const _Float16* __restrict__ Bt,
    const float* __restrict__ bias, float* __restrict__ Cf,
    _Float16* __restrict__ Ch, _Float16* __restrict__ Vh,
    int K, int lda, int ldbt, int ldcf, int ldch,
    long long sA, long long sB, long long sC,
    int gelu, int qkvmode, float scale)
{
    __shared__ alignas(16) _Float16 As[64][40];   // 80B row stride (16B multiple)
    __shared__ alignas(16) _Float16 Bs[128][40];

    const int tid  = threadIdx.x;
    const int lane = tid & 31;
    const int wave = tid >> 5;
    const int wm = wave >> 2;
    const int wn = wave & 3;
    const int row0 = blockIdx.y * 64;
    const int col0 = blockIdx.x * 128;
    const long long batch = blockIdx.z;
    A  += batch * sA;
    Bt += batch * sB;

    v8f acc[2][2] = {};
    const int ar = tid >> 2, ac = (tid & 3) << 3;        // A: 1 uint4 / thread
    const int brow = tid >> 1, bcol = (tid & 1) << 4;    // B: 2 uint4 / thread

    for (int k0 = 0; k0 < K; k0 += 32) {
        *(uint4*)&As[ar][ac]         = *(const uint4*)&A[(size_t)(row0 + ar) * lda + k0 + ac];
        *(uint4*)&Bs[brow][bcol]     = *(const uint4*)&Bt[(size_t)(col0 + brow) * ldbt + k0 + bcol];
        *(uint4*)&Bs[brow][bcol + 8] = *(const uint4*)&Bt[(size_t)(col0 + brow) * ldbt + k0 + bcol + 8];
        if (k0 + 32 < K)
            __builtin_prefetch(&A[(size_t)(row0 + ar) * lda + k0 + 32 + ac], 0, 3);
        __syncthreads();

        const int mn   = lane & 15;
        const int base = (lane >> 4) << 3;
        v16h af[2], bf[2];
        #pragma unroll
        for (int f = 0; f < 2; ++f) {
            af[f] = frag_ld(&As[wm * 32 + f * 16 + mn][0], base);
            bf[f] = frag_ld(&Bs[wn * 32 + f * 16 + mn][0], base);
        }
        #pragma unroll
        for (int fm = 0; fm < 2; ++fm)
            #pragma unroll
            for (int fn = 0; fn < 2; ++fn)
                acc[fm][fn] = __builtin_amdgcn_wmma_f32_16x16x32_f16(
                    false, af[fm], false, bf[fn], (short)0, acc[fm][fn], false, false);
        __syncthreads();
    }

    const int Nl  = lane & 15;
    const int Mhi = (lane >> 4) << 3;
    #pragma unroll
    for (int fm = 0; fm < 2; ++fm)
        #pragma unroll
        for (int fn = 0; fn < 2; ++fn)
            #pragma unroll
            for (int e = 0; e < 8; ++e) {
                int r = row0 + wm * 32 + fm * 16 + e + Mhi;
                int c = col0 + wn * 32 + fn * 16 + Nl;
                float v = acc[fm][fn][e] * scale;
                if (bias) v += bias[c];
                if (gelu) v = 0.5f * v * (1.0f + erff(v * 0.70710678118654752f));
                if (qkvmode) {
                    int sec = c >> 9, rem = c & 511;
                    int d = rem >> 3, hh = rem & 7;         // reference split: col = d*8 + h
                    if (sec < 2)
                        Ch[(size_t)r * 1024 + sec * 512 + hh * 64 + d] = (_Float16)v;
                    else
                        Vh[(((size_t)(r >> 10) * HEADS + hh) * DHEAD + d) * NN + (r & (NN - 1))] = (_Float16)v;
                } else {
                    if (Cf) Cf[batch * sC + (size_t)r * ldcf + c] = v;
                    if (Ch) Ch[batch * sC + (size_t)r * ldch + c] = (_Float16)v;
                }
            }
}

// ---------------------------------------------------------------------------
// Attention. qk: [4096][1024] f16 (cols 0-511 = Q at h*64+d, 512-1023 = K),
// vsep: [b][h][d][n] f16. One block = (b, h, 16 query rows), 8 waves.
// causal=1 masks j>i; causal=0 masks j<i. out fp32 merged heads [r][h*64+d].
// ---------------------------------------------------------------------------
__global__ __launch_bounds__(256) void attn_kernel(
    const _Float16* __restrict__ qk, const _Float16* __restrict__ vsep,
    float* __restrict__ out, int causal)
{
    const int qt  = blockIdx.x;
    const int h   = blockIdx.y;
    const int b   = blockIdx.z;
    const int tid = threadIdx.x;
    const int lane = tid & 31;
    const int wave = tid >> 5;
    const int qbase = qt * 16;
    const size_t rowbase = (size_t)b * NN;

    __shared__ alignas(16) _Float16 Sq[16][64];
    __shared__ alignas(16) _Float16 Skv[128][72];   // K block [j][d]; aliased as V^T
    __shared__ alignas(16) _Float16 Sp[16][1032];
    __shared__ float Sout[16][68];
    __shared__ float Sred[16][16];
    __shared__ float Srow[16];
    _Float16 (*Svt)[136] = (_Float16 (*)[136])Skv; // V^T [d][j] (64x136 <= 128x72)

    const float qscale = 0.044194173824159216f;    // 1/sqrt(512) (ref: /sqrt(planes))

    for (int idx = tid; idx < 16 * 64; idx += 256) {
        int r = idx >> 6, d = idx & 63;
        float qv = (float)qk[(rowbase + qbase + r) * 1024 + h * DHEAD + d];
        Sq[r][d] = (_Float16)(qv * qscale);
    }
    __syncthreads();

    const int mn   = lane & 15;
    const int base = (lane >> 4) << 3;
    v16h aq[2];
    #pragma unroll
    for (int c = 0; c < 2; ++c) aq[c] = frag_ld(&Sq[mn][c * 32], base);

    // ---- Phase 1: S = q.k^T + mask, f16 rows in Sp ----
    for (int jb = 0; jb < 8; ++jb) {
        for (int i4 = tid; i4 < 1024; i4 += 256) {          // 128x64 halves, uint4
            int r = i4 >> 3, c8 = (i4 & 7) << 3;
            *(uint4*)&Skv[r][c8] =
                *(const uint4*)&qk[(rowbase + jb * 128 + r) * 1024 + 512 + h * DHEAD + c8];
        }
        __syncthreads();
        const int jt = jb * 8 + wave;
        v8f c = {};
        #pragma unroll
        for (int ch = 0; ch < 2; ++ch) {
            v16h bf = frag_ld(&Skv[wave * 16 + mn][ch * 32], base);  // B = K^T (Bt = K)
            c = __builtin_amdgcn_wmma_f32_16x16x32_f16(
                false, aq[ch], false, bf, (short)0, c, false, false);
        }
        const int Mhi = (lane >> 4) << 3;
        #pragma unroll
        for (int e = 0; e < 8; ++e) {
            int i = qbase + e + Mhi;
            int j = jt * 16 + mn;
            float s = c[e];
            if (causal ? (j > i) : (j < i)) s = -INFINITY;
            Sp[e + Mhi][jt * 16 + mn] = (_Float16)s;
        }
        __syncthreads();
    }

    // ---- Phase 2: fp32 softmax over 1024 cols ----
    {
        const int r = tid >> 4, sub = tid & 15;
        float pm = -INFINITY;
        for (int c0 = sub; c0 < 1024; c0 += 16)
            pm = fmaxf(pm, (float)Sp[r][c0]);
        Sred[r][sub] = pm;
        __syncthreads();
        if (sub == 0) {
            float m_ = Sred[r][0];
            for (int q = 1; q < 16; ++q) m_ = fmaxf(m_, Sred[r][q]);
            Srow[r] = m_;
        }
        __syncthreads();
        const float rm = Srow[r];
        float ps = 0.f;
        for (int c0 = sub; c0 < 1024; c0 += 16) {
            float p = __expf((float)Sp[r][c0] - rm);
            Sp[r][c0] = (_Float16)p;
            ps += p;
        }
        Sred[r][sub] = ps;
        __syncthreads();
        if (sub == 0) {
            float s_ = 0.f;
            for (int q = 0; q < 16; ++q) s_ += Sred[r][q];
            Srow[r] = s_;
        }
        __syncthreads();
        const float inv = 1.0f / Srow[r];
        for (int c0 = sub; c0 < 1024; c0 += 16)
            Sp[r][c0] = (_Float16)((float)Sp[r][c0] * inv);
    }
    for (int idx = tid; idx < 16 * 64; idx += 256)
        Sout[idx >> 6][idx & 63] = 0.f;
    __syncthreads();

    // ---- Phase 3: out = P @ V; wave -> d-frag (w&3), j-chunk half (w>>2) ----
    {
        const int nfrag = wave & 3;
        const int chalf = wave >> 2;
        v8f o = {};
        for (int jb = 0; jb < 8; ++jb) {
            for (int i4 = tid; i4 < 1024; i4 += 256) {      // V^T block [d][j], uint4
                int d = i4 >> 4, c8 = (i4 & 15) << 3;
                *(uint4*)&Svt[d][c8] =
                    *(const uint4*)&vsep[(((size_t)b * HEADS + h) * DHEAD + d) * NN + jb * 128 + c8];
            }
            __syncthreads();
            for (int cc = chalf * 2; cc < chalf * 2 + 2; ++cc) {
                v16h af = frag_ld(&Sp[mn][jb * 128 + cc * 32], base);     // A = P
                v16h bf = frag_ld(&Svt[nfrag * 16 + mn][cc * 32], base);  // Bt = V^T
                o = __builtin_amdgcn_wmma_f32_16x16x32_f16(
                    false, af, false, bf, (short)0, o, false, false);
            }
            __syncthreads();
        }
        const int Mhi = (lane >> 4) << 3;
        #pragma unroll
        for (int e = 0; e < 8; ++e)
            atomicAdd(&Sout[e + Mhi][nfrag * 16 + mn], o[e]);  // ds_add_f32
    }
    __syncthreads();
    for (int idx = tid; idx < 16 * 64; idx += 256) {
        int r = idx >> 6, d = idx & 63;
        out[(rowbase + qbase + r) * PLANES + h * DHEAD + d] = Sout[r][d];
    }
}

// ---------------------------------------------------------------------------
// y = LayerNorm(x + r) * g + b ; one block per 512-wide row
// ---------------------------------------------------------------------------
__global__ __launch_bounds__(256) void resln_kernel(
    const float* __restrict__ x, const float* __restrict__ r,
    const float* __restrict__ g, const float* __restrict__ be,
    float* __restrict__ yf, _Float16* __restrict__ yh)
{
    __shared__ float red[256];
    __shared__ float red2[256];
    const int row = blockIdx.x;
    const int tid = threadIdx.x;
    const size_t o = (size_t)row * PLANES;
    float v0 = x[o + tid]       + r[o + tid];
    float v1 = x[o + tid + 256] + r[o + tid + 256];
    red[tid]  = v0 + v1;
    red2[tid] = v0 * v0 + v1 * v1;
    __syncthreads();
    for (int s = 128; s > 0; s >>= 1) {
        if (tid < s) { red[tid] += red[tid + s]; red2[tid] += red2[tid + s]; }
        __syncthreads();
    }
    const float mu   = red[0] * (1.0f / PLANES);
    const float var  = red2[0] * (1.0f / PLANES) - mu * mu;
    const float rstd = rsqrtf(var + 1e-5f);
    float y0 = (v0 - mu) * rstd * g[tid]       + be[tid];
    float y1 = (v1 - mu) * rstd * g[tid + 256] + be[tid + 256];
    if (yf) { yf[o + tid] = y0; yf[o + tid + 256] = y1; }
    if (yh) { yh[o + tid] = (_Float16)y0; yh[o + tid + 256] = (_Float16)y1; }
}

__global__ void sigmoid_kernel(const float* __restrict__ in,
                               _Float16* __restrict__ out) {
    int idx = blockIdx.x * blockDim.x + threadIdx.x;
    if (idx >= ROWS * PLANES) return;
    out[idx] = (_Float16)(1.0f / (1.0f + __expf(-in[idx])));
}

// ---------------------------------------------------------------------------
extern "C" void kernel_launch(void* const* d_in, const int* in_sizes, int n_in,
                              void* d_out, int out_size, void* d_ws, size_t ws_size,
                              hipStream_t stream) {
    (void)in_sizes; (void)n_in; (void)out_size; (void)ws_size;
    const float* x = (const float*)d_in[0];
    const float* wqkv[2] = {(const float*)d_in[1],  (const float*)d_in[11]};
    const float* bqkv[2] = {(const float*)d_in[2],  (const float*)d_in[12]};
    const float* ga[2]   = {(const float*)d_in[3],  (const float*)d_in[13]};
    const float* bea[2]  = {(const float*)d_in[4],  (const float*)d_in[14]};
    const float* w1[2]   = {(const float*)d_in[5],  (const float*)d_in[15]};
    const float* b1[2]   = {(const float*)d_in[6],  (const float*)d_in[16]};
    const float* w2[2]   = {(const float*)d_in[7],  (const float*)d_in[17]};
    const float* b2[2]   = {(const float*)d_in[8],  (const float*)d_in[18]};
    const float* gb[2]   = {(const float*)d_in[9],  (const float*)d_in[19]};
    const float* beb[2]  = {(const float*)d_in[10], (const float*)d_in[20]};

    char* ws = (char*)d_ws;
    size_t off = 0;
    auto alloc = [&](size_t bytes) {
        void* p = ws + off;
        off += (bytes + 255) & ~(size_t)255;
        return p;
    };
    _Float16* xh      = (_Float16*)alloc((size_t)ROWS * PLANES * 2);
    _Float16* wqkvt0  = (_Float16*)alloc((size_t)PLANES * QKVW * 2);   // [1536][512]
    _Float16* wqkvt1  = (_Float16*)alloc((size_t)PLANES * QKVW * 2);
    _Float16* w1t0    = (_Float16*)alloc((size_t)PLANES * PLANES * 2);
    _Float16* w1t1    = (_Float16*)alloc((size_t)PLANES * PLANES * 2);
    _Float16* w2t0    = (_Float16*)alloc((size_t)PLANES * PLANES * 2);
    _Float16* w2t1    = (_Float16*)alloc((size_t)PLANES * PLANES * 2);
    _Float16* qkh     = (_Float16*)alloc((size_t)ROWS * 1024 * 2);     // Q|K de-interleaved
    _Float16* vseph   = (_Float16*)alloc((size_t)BB * HEADS * DHEAD * NN * 2); // [b][h][d][n]
    float*    attnf   = (float*)   alloc((size_t)ROWS * PLANES * 4);
    float*    h1f     = (float*)   alloc((size_t)ROWS * PLANES * 4);
    _Float16* h1h     = (_Float16*)alloc((size_t)ROWS * PLANES * 2);
    _Float16* ffah    = (_Float16*)alloc((size_t)ROWS * PLANES * 2);
    float*    ffbf    = (float*)   alloc((size_t)ROWS * PLANES * 4);
    float*    h2f     = (float*)   alloc((size_t)ROWS * PLANES * 4);
    _Float16* td1h    = (_Float16*)alloc((size_t)ROWS * PLANES * 2);
    _Float16* td2h    = (_Float16*)alloc((size_t)ROWS * PLANES * 2);

    _Float16* wqkvt[2] = {wqkvt0, wqkvt1};
    _Float16* w1t[2]   = {w1t0, w1t1};
    _Float16* w2t[2]   = {w2t0, w2t1};

    cast_f16_kernel<<<(ROWS * PLANES + 255) / 256, 256, 0, stream>>>(x, xh, ROWS * PLANES);
    for (int L = 0; L < 2; ++L) {
        cast_transpose_kernel<<<(PLANES * QKVW + 255) / 256, 256, 0, stream>>>(
            wqkv[L], wqkvt[L], PLANES, QKVW);
        cast_transpose_kernel<<<(PLANES * PLANES + 255) / 256, 256, 0, stream>>>(
            w1[L], w1t[L], PLANES, PLANES);
        cast_transpose_kernel<<<(PLANES * PLANES + 255) / 256, 256, 0, stream>>>(
            w2[L], w2t[L], PLANES, PLANES);
    }

    for (int L = 0; L < 2; ++L) {
        // QKV projection + head de-interleave
        gemm_f16_kernel<<<dim3(QKVW / 128, ROWS / 64, 1), 256, 0, stream>>>(
            xh, wqkvt[L], bqkv[L], nullptr, qkh, vseph,
            PLANES, PLANES, PLANES, 0, 0, 0, 0, 0, /*gelu=*/0, /*qkvmode=*/1, 1.0f);
        attn_kernel<<<dim3(NN / 16, HEADS, BB), 256, 0, stream>>>(
            qkh, vseph, attnf, L == 0 ? 1 : 0);
        resln_kernel<<<ROWS, 256, 0, stream>>>(x, attnf, ga[L], bea[L], h1f, h1h);
        // FFN1 + exact GELU
        gemm_f16_kernel<<<dim3(PLANES / 128, ROWS / 64, 1), 256, 0, stream>>>(
            h1h, w1t[L], b1[L], nullptr, ffah, nullptr,
            PLANES, PLANES, PLANES, 0, PLANES, 0, 0, 0, /*gelu=*/1, 0, 1.0f);
        // FFN2
        gemm_f16_kernel<<<dim3(PLANES / 128, ROWS / 64, 1), 256, 0, stream>>>(
            ffah, w2t[L], b2[L], ffbf, nullptr, nullptr,
            PLANES, PLANES, PLANES, PLANES, 0, 0, 0, 0, /*gelu=*/0, 0, 1.0f);
        resln_kernel<<<ROWS, 256, 0, stream>>>(h1f, ffbf, gb[L], beb[L], h2f, nullptr);
        sigmoid_kernel<<<(ROWS * PLANES + 255) / 256, 256, 0, stream>>>(
            h2f, (L == 0) ? td1h : td2h);
    }

    // sim[b] = td1[b] @ td2[b]^T / 512 : Bt = td2 (natural layout), batched
    gemm_f16_kernel<<<dim3(NN / 128, NN / 64, BB), 256, 0, stream>>>(
        td1h, td2h, nullptr, (float*)d_out, nullptr, nullptr,
        PLANES, PLANES, PLANES, NN, 0,
        (long long)NN * PLANES, (long long)NN * PLANES, (long long)NN * NN,
        /*gelu=*/0, 0, 1.0f / (float)PLANES);
}
